// GraphAttentionLayer_31421980737995
// MI455X (gfx1250) — compile-verified, matrix operand order checked by reading, not run on hardware
//
#include <hip/hip_runtime.h>
#include <hip/hip_bf16.h>

// GAT layer for MI455X (gfx1250), wave32 + WMMA bf16 + TDM async tensor loads.
// HBM floor: one pass over adj (151MB) ~6.5us @ 23.3TB/s. The per-(i,j,h) logit
// work is VALU-bound: logits computed in float2 so the backend can form packed
// VOP3P f32 ops (v_pk_add/mul/fma_f32); softmax row-sums are folded into an
// extra WMMA against a ones-matrix (free on the matrix pipe); adj tiles are
// staged by the Tensor Data Mover with padding producing the bank-conflict-free
// LDS layout directly.

#define NN   6144
#define INF_ 512
#define HF   512
#define H    8
#define F    64

typedef __attribute__((ext_vector_type(16))) __bf16          v16bf;
typedef __attribute__((ext_vector_type(2)))  __bf16          v2bf;
typedef __attribute__((ext_vector_type(16))) unsigned short  v16u;
typedef __attribute__((ext_vector_type(8)))  float           v8f;
typedef __attribute__((ext_vector_type(2)))  float           v2f;
typedef __attribute__((ext_vector_type(4)))  unsigned int    v4u;
typedef __attribute__((ext_vector_type(4)))  int             v4i;
typedef __attribute__((ext_vector_type(8)))  int             v8i;

#if defined(__has_builtin)
#if __has_builtin(__builtin_amdgcn_cvt_pk_bf16_f32)
#define HAVE_PK 1
#endif
#if __has_builtin(__builtin_amdgcn_tensor_load_to_lds) && \
    __has_builtin(__builtin_amdgcn_s_wait_tensorcnt)
#define USE_TDM 1
#endif
#endif
#ifndef HAVE_PK
#define HAVE_PK 0
#endif
#ifndef USE_TDM
#define USE_TDM 0
#endif

#define LOG2E 1.4426950408889634f
#define BIG2  1.5e10f   // ~1e10 * log2(e): anything huge works (exp2 -> 0)

__device__ __forceinline__ unsigned short f2bf(float f) {
    union { __bf16 h; unsigned short s; } u;
    u.h = (__bf16)f;                       // backend picks native cvt if present
    return u.s;
}
__device__ __forceinline__ float bf2f(unsigned short s) {
    union { __bf16 h; unsigned short s; } u;
    u.s = s;
    return (float)u.h;
}
__device__ __forceinline__ unsigned int pack_bf16(float a, float b) {
#if HAVE_PK
    union { v2bf p; unsigned int w; } u;
    u.p = __builtin_amdgcn_cvt_pk_bf16_f32(a, b);
    return u.w;
#else
    union { v2bf p; unsigned int w; } u;
    u.p[0] = (__bf16)a; u.p[1] = (__bf16)b;
    return u.w;
#endif
}

union BfVec { v16u u; v16bf b; unsigned int d[8]; };

// ---------------------------------------------------------------------------
// K0: weight[k][n] (f32) -> wT[n][k] (bf16) so WMMA B-tiles are contiguous.
__global__ __launch_bounds__(256) void k_transpose_w(const float* __restrict__ w,
                                                     unsigned short* __restrict__ wT) {
    int id = blockIdx.x * 256 + threadIdx.x;      // 512*512
    int n = id >> 9, k = id & 511;
    wT[id] = f2bf(w[(size_t)k * HF + n]);
}

// ---------------------------------------------------------------------------
// K1: whT[f][n] (bf16) = (x @ W)^T via WMMA bf16.  One wave per 16m x 64n tile.
__global__ __launch_bounds__(32) void k_gemm_wh(const float* __restrict__ x,
                                                const unsigned short* __restrict__ wT,
                                                unsigned short* __restrict__ whT) {
    const int lane  = threadIdx.x;
    const int m     = lane & 15;
    const int khalf = lane >> 4;          // 0/1 -> contiguous K half (k = 16*khalf + s) in A and B
    const int Klo   = khalf * 16;
    const int m0    = (blockIdx.x >> 3) * 16;
    const int n0    = (blockIdx.x & 7) * 64;

    v8f acc[4];
#pragma unroll
    for (int c = 0; c < 4; ++c) acc[c] = (v8f)(0.0f);

    for (int k = 0; k < INF_; k += 32) {
        // Issue all B loads first so the scheduler can wait progressively.
        BfVec B[4];
#pragma unroll
        for (int c = 0; c < 4; ++c)
            B[c].u = *(const v16u*)&wT[(size_t)(n0 + c * 16 + m) * INF_ + k + Klo];
        BfVec A;
        const float* xp = &x[(size_t)(m0 + m) * INF_ + k + Klo];
#pragma unroll
        for (int v = 0; v < 8; ++v) A.d[v] = pack_bf16(xp[2 * v], xp[2 * v + 1]);
#pragma unroll
        for (int c = 0; c < 4; ++c)
            acc[c] = __builtin_amdgcn_wmma_f32_16x16x32_bf16(
                false, A.b, false, B[c].b, (short)0, acc[c], false, false);
    }
    // D layout: VGPR r holds row M = r + 8*khalf, col N = m.  Rows r,r+1 are
    // consecutive n in whT -> store packed b32 pairs.
#pragma unroll
    for (int c = 0; c < 4; ++c)
#pragma unroll
        for (int q = 0; q < 4; ++q) {
            unsigned int w = pack_bf16(acc[c][2 * q], acc[c][2 * q + 1]);
            *(unsigned int*)&whT[(size_t)(n0 + c * 16 + m) * NN + m0 + 2 * q + 8 * khalf] = w;
        }
}

// ---------------------------------------------------------------------------
// K2: awh_i[h][n] = log2(e) * sum_f wh[n,h,f]*li[h,f]  (pre-scaled so the main
// kernel can use native exp2); same for awh_j.
__global__ __launch_bounds__(256) void k_awh(const unsigned short* __restrict__ whT,
                                             const float* __restrict__ li,
                                             const float* __restrict__ lj,
                                             float* __restrict__ awh_i,
                                             float* __restrict__ awh_j) {
    int id = blockIdx.x * 256 + threadIdx.x;      // H*NN
    int h = id / NN, n = id - h * NN;
    float si = 0.f, sj = 0.f;
#pragma unroll 8
    for (int f = 0; f < F; ++f) {
        float w = bf2f(whT[(size_t)(h * F + f) * NN + n]);
        si += w * li[h * F + f];
        sj += w * lj[h * F + f];
    }
    awh_i[id] = si * LOG2E;
    awh_j[id] = sj * LOG2E;
}

// ---------------------------------------------------------------------------
// K3: fused attention.  Block = 32 output rows, 8 waves (1 wave/head).
// adj streamed once, staged to LDS via TDM (tensor_load_to_lds + TENSORcnt);
// p-tiles built in WMMA A layout; row-sums via WMMA against a ones matrix.
__global__ __launch_bounds__(256) void k_gat_attn(const float* __restrict__ adj,
                                                  const unsigned short* __restrict__ whT,
                                                  const float* __restrict__ awh_i,
                                                  const float* __restrict__ awh_j,
                                                  float* __restrict__ out) {
    __shared__ float s_adj[32][34];   // stride 34: 8B-aligned rows, conflict-free
    __shared__ float s_aj[H][32];

    const int tid   = threadIdx.x;
    const int lane  = tid & 31;
    const int h     = tid >> 5;           // head
    const int m     = lane & 15;          // row within 16-tile / N column
    const int khalf = lane >> 4;
    const int Klo   = khalf * 16;
    const int i0    = blockIdx.x * 32;

    const float ai0 = awh_i[h * NN + i0 + m];
    const float ai1 = awh_i[h * NN + i0 + 16 + m];
    const v2f ai2[2] = { { ai0, ai0 }, { ai1, ai1 } };

    v8f acc[2][4], accden[2];
#pragma unroll
    for (int t = 0; t < 2; ++t) {
        accden[t] = (v8f)(0.0f);
#pragma unroll
        for (int c = 0; c < 4; ++c) acc[t][c] = (v8f)(0.0f);
    }

    BfVec Bones;                           // 32x16 of bf16(1.0): row-sum matrix
#pragma unroll
    for (int v = 0; v < 8; ++v) Bones.d[v] = 0x3F803F80u;

    const int ar = tid >> 3;               // fallback adj stage: row 0..31
    const int ac = (tid & 7) * 4;

#if USE_TDM
    // D# group1 is loop-invariant: 2D f32 tile 32x32, tensor row len/stride 6144,
    // LDS pad: +2 DWORDs every 32 DWORDs -> [32][34] layout.
    const v8i g1 = { 0x03120000,            // data_size=4B, pad_en, intv=32dw, amt=2dw
                     0x18000000,            // tensor_dim0 = 6144 (low16 in [31:16])
                     0x18000000,            // dim0 hi=0 | tensor_dim1=6144 low16
                     0x00200000,            // dim1 hi=0 | tile_dim0=32
                     32,                    // tile_dim1=32, tile_dim2=0
                     6144,                  // tensor_dim0_stride lo
                     0, 0 };
    const v4i gz4 = { 0, 0, 0, 0 };
    const v8i gz8 = { 0, 0, 0, 0, 0, 0, 0, 0 };
    const unsigned int lds_base = (unsigned int)(size_t)(void*)&s_adj[0][0];
#endif

    const v2f c02  = (v2f)(0.2f);
    const v2f cBIG = (v2f)(BIG2);

    for (int jb = 0; jb < NN; jb += 32) {
#if USE_TDM
        if (h == 0) {                      // wave 0 drives the Tensor Data Mover
            unsigned long long ga =
                (unsigned long long)(const void*)&adj[(size_t)i0 * NN + jb];
            v4u g0;
            g0[0] = 1u;                    // count=1 (valid user descriptor)
            g0[1] = lds_base;              // lds_addr
            g0[2] = (unsigned int)ga;      // global_addr[31:0]
            g0[3] = (unsigned int)((ga >> 32) & 0x01FFFFFFu) | (2u << 30); // type=2
            __builtin_amdgcn_tensor_load_to_lds(g0, g1, gz4, gz4, gz8, 0);
        }
        s_aj[tid >> 5][tid & 31] = awh_j[(size_t)(tid >> 5) * NN + jb + (tid & 31)];
        if (h == 0) __builtin_amdgcn_s_wait_tensorcnt(0);
#else
        float4 a4 = *(const float4*)&adj[(size_t)(i0 + ar) * NN + jb + ac];
        s_adj[ar][ac + 0] = a4.x; s_adj[ar][ac + 1] = a4.y;
        s_adj[ar][ac + 2] = a4.z; s_adj[ar][ac + 3] = a4.w;
        s_aj[tid >> 5][tid & 31] = awh_j[(size_t)(tid >> 5) * NN + jb + (tid & 31)];
#endif
        __syncthreads();

        // prefetch next adj tile into GL2 while we compute
        if (jb + 32 < NN)
            __builtin_prefetch(&adj[(size_t)(i0 + ar) * NN + jb + 32 + ac], 0, 1);

        // ---- B tiles: wh[j=K, f=N] for this head (contiguous from whT) ----
        BfVec B[4];
#pragma unroll
        for (int c = 0; c < 4; ++c)
            B[c].u = *(const v16u*)&whT[(size_t)(h * F + c * 16 + m) * NN + jb + Klo];

        v2f aj2[8];
        const v2f* s_aj2 = (const v2f*)&s_aj[h][Klo];       // 8B-aligned
#pragma unroll
        for (int q = 0; q < 8; ++q) aj2[q] = s_aj2[q];

        // ---- p tiles in WMMA A layout: lane = row m, K = Klo..Klo+15 ----
        BfVec A[2];
#pragma unroll
        for (int t = 0; t < 2; ++t) {
            const v2f* arow = (const v2f*)&s_adj[t * 16 + m][Klo];  // 8B-aligned
#pragma unroll
            for (int q = 0; q < 8; ++q) {
                v2f s = ai2[t] + aj2[q];                            // pk_add
                v2f l = __builtin_elementwise_max(s, c02 * s);      // lrelu
                v2f z = __builtin_elementwise_fma(cBIG, arow[q], l - cBIG);
                float p0 = __builtin_amdgcn_exp2f(z[0]);            // masked -> 0
                float p1 = __builtin_amdgcn_exp2f(z[1]);
                A[t].d[q] = pack_bf16(p0, p1);
            }
        }
#pragma unroll
        for (int t = 0; t < 2; ++t) {
#pragma unroll
            for (int c = 0; c < 4; ++c)
                acc[t][c] = __builtin_amdgcn_wmma_f32_16x16x32_bf16(
                    false, A[t].b, false, B[c].b, (short)0, acc[t][c], false, false);
            accden[t] = __builtin_amdgcn_wmma_f32_16x16x32_bf16(
                false, A[t].b, false, Bones.b, (short)0, accden[t], false, false);
        }
        __syncthreads();
    }

    // D-layout rows of accden match output rows exactly: no shuffle needed.
#pragma unroll
    for (int t = 0; t < 2; ++t)
#pragma unroll
        for (int r = 0; r < 8; ++r) {
            const int row = t * 16 + r + 8 * khalf;
            const float inv = 1.0f / accden[t][r];
#pragma unroll
            for (int c = 0; c < 4; ++c)
                out[(size_t)(i0 + row) * HF + h * F + c * 16 + m] = acc[t][c][r] * inv;
        }
}

// ---------------------------------------------------------------------------
extern "C" void kernel_launch(void* const* d_in, const int* in_sizes, int n_in,
                              void* d_out, int out_size, void* d_ws, size_t ws_size,
                              hipStream_t stream) {
    const float* x      = (const float*)d_in[0];   // [6144,512]
    const float* adj    = (const float*)d_in[1];   // [6144,6144]
    const float* weight = (const float*)d_in[2];   // [512,512]
    const float* li     = (const float*)d_in[3];   // [1,8,64]
    const float* lj     = (const float*)d_in[4];   // [1,8,64]

    char* ws = (char*)d_ws;
    unsigned short* whT = (unsigned short*)ws;                         // 512*6144 bf16
    unsigned short* wT  = (unsigned short*)(ws + (size_t)HF * NN * 2); // 512*512 bf16
    float* awh_i = (float*)(ws + (size_t)HF * NN * 2 + (size_t)INF_ * HF * 2);
    float* awh_j = awh_i + (size_t)H * NN;

    k_transpose_w<<<(INF_ * HF) / 256, 256, 0, stream>>>(weight, wT);
    k_gemm_wh<<<(NN / 16) * (HF / 64), 32, 0, stream>>>(x, wT, whT);
    k_awh<<<(H * NN) / 256, 256, 0, stream>>>(whT, li, lj, awh_i, awh_j);
    k_gat_attn<<<NN / 32, 256, 0, stream>>>(adj, whT, awh_i, awh_j, (float*)d_out);
}
